// LinearAttention3_8589934685
// MI455X (gfx1250) — compile-verified, matrix operand order checked by reading, not run on hardware
//
#include <hip/hip_runtime.h>
#include <cstddef>

typedef __attribute__((ext_vector_type(16))) _Float16 v16h;
typedef __attribute__((ext_vector_type(8)))  float    v8f;

#define BATCH 16
#define CCH   256
#define HD    128
#define HH    48
#define NN    2304   // 48*48

// ---- fragment load helpers (CDNA5 16x16x32 f16 layout) --------------------
// A: 16(M)x32(K). lane<16: row M=lane, K = {0..7}U{16..23}; lane>=16: row
// M=lane-16, K = {8..15}U{24..31}.  p points at (row, k0+kb); runs contiguous.
__device__ __forceinline__ v16h frag_a_f16(const _Float16* p) {
    v16h a;
#pragma unroll
    for (int e = 0; e < 8; ++e) { a[e] = p[e]; a[e + 8] = p[16 + e]; }
    return a;
}
__device__ __forceinline__ v16h frag_a_f32(const float* p) {
    v16h a;
#pragma unroll
    for (int e = 0; e < 8; ++e) {
        a[e]     = (_Float16)p[e];
        a[e + 8] = (_Float16)p[16 + e];
    }
    return a;
}
// B: 32(K)x16(N), lane column N=lane&15, same K split. p at (k0+kb, col),
// consecutive K rows `stride` apart (stride==1 when K is the fastest dim).
__device__ __forceinline__ v16h frag_b_f32(const float* p, size_t stride) {
    v16h bm;
#pragma unroll
    for (int e = 0; e < 8; ++e) {
        bm[e]     = (_Float16)p[(size_t)e * stride];
        bm[e + 8] = (_Float16)p[(size_t)(16 + e) * stride];
    }
    return bm;
}

// ---------------------------------------------------------------------------
// K1: qk[o][n] = elu( sum_c W[o][c]*x[b][c][n] + b[o] ) + 1
//     o in [0,128)   -> qT[b][n][o]  (f16, transposed for K3 A-frags)
//     o in [128,256) -> k [b][o-128][n]
// One wave: 16(o) x 64(n), A fragment reused across 4 WMMAs per K-step.
// ---------------------------------------------------------------------------
__global__ void la3_k1_qk(const float* __restrict__ x,
                          const float* __restrict__ wqk,
                          const float* __restrict__ bqk,
                          _Float16* __restrict__ qT,
                          _Float16* __restrict__ kbuf)
{
    const int b    = blockIdx.y;
    const int tile = blockIdx.x * 8 + (threadIdx.x >> 5);   // 0..575
    const int o0   = (tile / 36) * 16;                      // 16 o-tiles
    const int n0   = (tile % 36) * 64;                      // 36 n4-tiles
    const int lane = threadIdx.x & 31;
    const int lr   = lane & 15;
    const int kb   = (lane & 16) ? 8 : 0;

    const float* xb = x + (size_t)b * CCH * NN;
    v8f acc[4] = {};
    for (int k0 = 0; k0 < CCH; k0 += 32) {
        const v16h a = frag_a_f32(wqk + (size_t)(o0 + lr) * CCH + k0 + kb);
        const float* xcol = xb + (size_t)(k0 + kb) * NN + n0 + lr;
#pragma unroll
        for (int t = 0; t < 4; ++t) {
            const v16h bm = frag_b_f32(xcol + 16 * t, NN);
            acc[t] = __builtin_amdgcn_wmma_f32_16x16x32_f16(
                         false, a, false, bm, (short)0, acc[t], false, false);
        }
    }

    const int mb = (lane & 16) ? 8 : 0;
    if (o0 < HD) {            // q half -> [b][n][d], contiguous in j
#pragma unroll
        for (int t = 0; t < 4; ++t) {
            const int n = n0 + 16 * t + lr;
            _Float16* qp = qT + ((size_t)b * NN + n) * HD + o0 + mb;
#pragma unroll
            for (int j = 0; j < 8; ++j) {
                float v = acc[t][j] + bqk[o0 + mb + j];
                v = (v > 0.0f) ? (v + 1.0f) : __expf(v);
                qp[j] = (_Float16)v;
            }
        }
    } else {                  // k half -> [b][d][m]
#pragma unroll
        for (int t = 0; t < 4; ++t) {
            const int n = n0 + 16 * t + lr;
#pragma unroll
            for (int j = 0; j < 8; ++j) {
                const int o = o0 + mb + j;
                float v = acc[t][j] + bqk[o];
                v = (v > 0.0f) ? (v + 1.0f) : __expf(v);
                kbuf[((size_t)b * HD + (o - HD)) * NN + n] = (_Float16)v;
            }
        }
    }
}

// ---------------------------------------------------------------------------
// K2: KV[d][c] = sum_m k[d][m] * x[b][c][m]   (128 x 256, K = 2304)
// One wave: 16(d) x 64(c), 72 K-steps x 4 WMMAs, A reused 4x. All fragment
// loads are contiguous runs (K == fastest dim of both operands).
// ---------------------------------------------------------------------------
__global__ void la3_k2_kv(const float* __restrict__ x,
                          const _Float16* __restrict__ kbuf,
                          float* __restrict__ kv)
{
    const int b    = blockIdx.y;
    const int tile = blockIdx.x * 8 + (threadIdx.x >> 5);   // 0..31
    const int d0   = (tile / 4) * 16;                       // 8 d-tiles
    const int c4   = (tile % 4) * 64;                       // 4 c4-tiles
    const int lane = threadIdx.x & 31;
    const int lr   = lane & 15;
    const int kb   = (lane & 16) ? 8 : 0;

    const _Float16* kp = kbuf + (size_t)b * HD * NN;
    const float*    xb = x    + (size_t)b * CCH * NN;
    v8f acc[4] = {};
    for (int k0 = 0; k0 < NN; k0 += 32) {
        const v16h a = frag_a_f16(kp + (size_t)(d0 + lr) * NN + k0 + kb);
#pragma unroll
        for (int t = 0; t < 4; ++t) {
            const v16h bm = frag_b_f32(xb + (size_t)(c4 + 16 * t + lr) * NN + k0 + kb, 1);
            acc[t] = __builtin_amdgcn_wmma_f32_16x16x32_f16(
                         false, a, false, bm, (short)0, acc[t], false, false);
        }
    }
    const int mb = (lane & 16) ? 8 : 0;
#pragma unroll
    for (int t = 0; t < 4; ++t)
#pragma unroll
        for (int j = 0; j < 8; ++j)
            kv[((size_t)b * HD + d0 + mb + j) * CCH + c4 + 16 * t + lr] = acc[t][j];
}

// ksum[b][d] = sum_m k[b][d][m]
__global__ void la3_k2_ksum(const _Float16* __restrict__ kbuf,
                            float* __restrict__ ksum)
{
    const int b = blockIdx.x, d = threadIdx.x;
    const _Float16* kr = kbuf + ((size_t)b * HD + d) * NN;
    float s = 0.0f;
    for (int m = 0; m < NN; ++m) s += (float)kr[m];
    ksum[b * HD + d] = s;
}

// denom[b][n] = (q_n . ksum)/N + 1e-6
__global__ void la3_k2_denom(const _Float16* __restrict__ qT,
                             const float* __restrict__ ksum,
                             float* __restrict__ denom)
{
    const int b = blockIdx.y;
    const int n = blockIdx.x * 256 + threadIdx.x;
    const _Float16* qp = qT + ((size_t)b * NN + n) * HD;
    const float*    ks = ksum + b * HD;
    float s = 0.0f;
#pragma unroll 8
    for (int d = 0; d < HD; ++d) s += (float)qp[d] * ks[d];
    denom[b * NN + n] = s * (1.0f / (float)NN) + 1e-6f;
}

// ---------------------------------------------------------------------------
// K3: out[b][c][n] = (q_n . KV[:,c]) * s^2 / denom[n]  + depthwise3x3(x)+b_pe
// One wave: 16(n) x 64(c), K = 128 (4 K-steps x 4 WMMAs, A reused 4x).
// Epilogue: each lane owns 4 channels x one contiguous 1x8 n-strip
// (16 | 48 -> strips never cross an image row); conv fused, stores coalesced.
// ---------------------------------------------------------------------------
__global__ void la3_k3_out(const float* __restrict__ x,
                           const _Float16* __restrict__ qT,
                           const float* __restrict__ kv,
                           const float* __restrict__ denom,
                           const float* __restrict__ wpe,
                           const float* __restrict__ bpe,
                           float* __restrict__ out)
{
    const int b    = blockIdx.y;
    const int tile = blockIdx.x * 8 + (threadIdx.x >> 5);   // 0..575
    const int n0   = (tile / 4) * 16;                       // 144 n-tiles
    const int c4   = (tile % 4) * 64;                       // 4 c4-tiles
    const int lane = threadIdx.x & 31;
    const int lr   = lane & 15;
    const int kb   = (lane & 16) ? 8 : 0;

    const _Float16* qb  = qT + (size_t)b * NN * HD;
    const float*    kvb = kv + (size_t)b * HD * CCH;
    v8f acc[4] = {};
#pragma unroll
    for (int k0 = 0; k0 < HD; k0 += 32) {
        const v16h a = frag_a_f16(qb + (size_t)(n0 + lr) * HD + k0 + kb);
        const float* kvrow = kvb + (size_t)(k0 + kb) * CCH + c4 + lr;
#pragma unroll
        for (int t = 0; t < 4; ++t) {
            const v16h bm = frag_b_f32(kvrow + 16 * t, CCH);
            acc[t] = __builtin_amdgcn_wmma_f32_16x16x32_f16(
                         false, a, false, bm, (short)0, acc[t], false, false);
        }
    }

    const int mb     = (lane & 16) ? 8 : 0;
    const int nstart = n0 + mb;              // 8 consecutive n, one image row
    const int y      = nstart / HH;
    const int xs     = nstart % HH;

    // per-n scale hoisted out of the channel loop
    const float s2 = 1.0f / (float)NN;
    float inv[8];
    {
        const float* dn = denom + b * NN + nstart;
#pragma unroll
        for (int j = 0; j < 8; ++j) inv[j] = s2 / dn[j];
    }

#pragma unroll
    for (int t = 0; t < 4; ++t) {
        const int c = c4 + 16 * t + lr;
        float w9[9];
#pragma unroll
        for (int i = 0; i < 9; ++i) w9[i] = wpe[c * 9 + i];

        const float* xc = x + ((size_t)b * CCH + c) * NN;
        float rowbuf[3][10];
#pragma unroll
        for (int dy = 0; dy < 3; ++dy) {
            const int yy = y + dy - 1;
#pragma unroll
            for (int dx = 0; dx < 10; ++dx) {
                const int xx = xs + dx - 1;
                rowbuf[dy][dx] = (yy >= 0 && yy < HH && xx >= 0 && xx < HH)
                                   ? xc[yy * HH + xx] : 0.0f;
            }
        }

        const float bias = bpe[c];
        float* ob = out + ((size_t)b * CCH + c) * NN + nstart;
#pragma unroll
        for (int j = 0; j < 8; ++j) {
            float pe = bias;
#pragma unroll
            for (int dy = 0; dy < 3; ++dy)
#pragma unroll
                for (int dx = 0; dx < 3; ++dx)
                    pe += w9[dy * 3 + dx] * rowbuf[dy][j + dx];
            ob[j] = acc[t][j] * inv[j] + pe;
        }
    }
}

// ---------------------------------------------------------------------------
extern "C" void kernel_launch(void* const* d_in, const int* in_sizes, int n_in,
                              void* d_out, int out_size, void* d_ws, size_t ws_size,
                              hipStream_t stream)
{
    (void)in_sizes; (void)n_in; (void)out_size; (void)ws_size;
    const float* x   = (const float*)d_in[0];
    const float* wqk = (const float*)d_in[1];
    const float* bqk = (const float*)d_in[2];
    const float* wpe = (const float*)d_in[3];
    const float* bpe = (const float*)d_in[4];
    float* out = (float*)d_out;

    char* ws = (char*)d_ws;
    const size_t qT_bytes   = (size_t)BATCH * NN * HD * sizeof(_Float16);   // 9.44 MB
    const size_t k_bytes    = (size_t)BATCH * HD * NN * sizeof(_Float16);   // 9.44 MB
    const size_t kv_bytes   = (size_t)BATCH * HD * CCH * sizeof(float);     // 2.10 MB
    const size_t ksum_bytes = (size_t)BATCH * HD * sizeof(float);           // 8 KB
    _Float16* qT   = (_Float16*)(ws);
    _Float16* kbuf = (_Float16*)(ws + qT_bytes);
    float*    kv   = (float*)(ws + qT_bytes + k_bytes);
    float*    ksum = (float*)(ws + qT_bytes + k_bytes + kv_bytes);
    float*    den  = (float*)(ws + qT_bytes + k_bytes + kv_bytes + ksum_bytes);

    const dim3 blk(256);
    la3_k1_qk   <<<dim3(72, BATCH), blk,       0, stream>>>(x, wqk, bqk, qT, kbuf);
    la3_k2_kv   <<<dim3(4,  BATCH), blk,       0, stream>>>(x, kbuf, kv);
    la3_k2_ksum <<<dim3(BATCH),     dim3(128), 0, stream>>>(kbuf, ksum);
    la3_k2_denom<<<dim3(9,  BATCH), blk,       0, stream>>>(qT, ksum, den);
    la3_k3_out  <<<dim3(72, BATCH), blk,       0, stream>>>(x, qT, kv, den, wpe, bpe, out);
}
